// VanillaRNN_87909390615007
// MI455X (gfx1250) — compile-verified, matrix operand order checked by reading, not run on hardware
//
#include <hip/hip_runtime.h>
#include <hip/hip_bf16.h>

// ---------------------------------------------------------------------------
// Fused VanillaRNN for gfx1250 (MI455X), wave32 + WMMA bf16.
//
//   B=512, T=1024, D_IN=64, H=256, C=128
//   out = (scan_t h = tanh(x_t@Wxh + bh + h@Whh)) @ Whp + bp
//
// Batch rows are independent through the scan: 32 workgroups x 16 batch rows
// (one WMMA M-tile each), h double-buffered in LDS (bf16), 16 waves per WG
// each owning a 16-wide N-tile of H with its [320x16] slice of [Whh;Wxh]
// resident in VGPRs as WMMA-B operands.  One s_barrier per timestep, x_t
// streamed from HBM (non-temporal) exactly once, fused into the K dimension.
// tanh uses the CDNA5 hardware transcendental (v_tanh_f32) when the builtin
// exists, else a branch-free exp2/rcp expansion.
// ---------------------------------------------------------------------------

typedef __bf16 bf16_t;
typedef bf16_t v16bf __attribute__((ext_vector_type(16)));
typedef bf16_t v8bf  __attribute__((ext_vector_type(8)));
typedef float  v8f   __attribute__((ext_vector_type(8)));

namespace {
constexpr int Tn   = 1024;
constexpr int DIN  = 64;
constexpr int H    = 256;
constexpr int C    = 128;
constexpr int KTOT = H + DIN;        // 320 : [h | x_t] concatenated K
constexpr int NK   = KTOT / 32;      // 10 WMMA K-chunks per step
constexpr int MROW = 16;             // batch rows per workgroup (WMMA M)
constexpr int STR  = KTOT + 8;       // LDS row stride in bf16 (656B = 41*16)
}

union V16 { v16bf v; v8bf h[2]; bf16_t e[16]; };

// Branch-free tanh: hardware v_tanh_f32 if clang exposes it, else exp2+rcp.
static __device__ inline float fast_tanh(float v) {
#if __has_builtin(__builtin_amdgcn_tanhf)
    return __builtin_amdgcn_tanhf(v);
#elif __has_builtin(__builtin_amdgcn_tanh_f32)
    return __builtin_amdgcn_tanh_f32(v);
#else
    // tanh(x) = (1 - e) / (1 + e),  e = exp(-2x) = exp2(-2*log2(e)*x)
    float xc = fminf(fmaxf(v, -20.0f), 20.0f);            // range-safe
    float e  = __builtin_amdgcn_exp2f(xc * -2.8853900817779268f);
    return (1.0f - e) * __builtin_amdgcn_rcpf(1.0f + e);
#endif
}

__global__ __launch_bounds__(512, 1)
void VanillaRNN_fused_kernel(const float* __restrict__ x,
                             const float* __restrict__ Wxh,
                             const float* __restrict__ Whh,
                             const float* __restrict__ bh,
                             const float* __restrict__ Whp,
                             const float* __restrict__ bp,
                             float* __restrict__ out)
{
    __shared__ __align__(16) bf16_t hbuf[2][MROW][STR];   // [h(256) | x(64) | pad]

    const int tid  = threadIdx.x;
    const int lane = tid & 31;
    const int wave = tid >> 5;              // 0..15 : N-tile of H
    const int wg   = blockIdx.x;            // 0..31 : M-tile (batch rows)
    const int nlo  = lane & 15;             // N within tile == A-row for loads
    const int khi  = (lane >> 4) << 3;      // +8 K offset for lanes 16..31

    // ---- one-time: load this wave's [320x16] weight slice as WMMA-B regs ----
    // B-operand (32x16, bf16): lane L holds column N=L%16; element e maps to
    // K = 32c + khi + (e<8 ? e : e-8+16)  (mirror of the documented A layout).
    V16 w[NK];
    const int ncol = wave * 16 + nlo;
    #pragma unroll
    for (int c = 0; c < NK; ++c) {
        #pragma unroll
        for (int e = 0; e < 16; ++e) {
            const int k = c * 32 + khi + (e < 8 ? e : (e - 8) + 16);
            const float f = (k < H) ? Whh[(size_t)k * H + ncol]
                                    : Wxh[(size_t)(k - H) * H + ncol];
            w[c].e[e] = (bf16_t)f;
        }
    }
    const float bias = bh[ncol];

    // ---- init: h0 = 0, stage x(t=0) into buffer 0 ----
    for (int i = tid; i < MROW * H; i += 512)
        hbuf[0][i >> 8][i & (H - 1)] = (bf16_t)0.0f;
    #pragma unroll
    for (int rep = 0; rep < 2; ++rep) {
        const int idx = tid + rep * 512;            // 1024 elems = 16 x 64
        const int r = idx >> 6, col = idx & 63;
        const float f = __builtin_nontemporal_load(
            x + (size_t)(wg * MROW + r) * Tn * DIN + col);
        hbuf[0][r][H + col] = (bf16_t)f;
    }
    __syncthreads();

    // ---- time loop: h_{t+1} = tanh(bh + [h_t|x_t] @ [Whh;Wxh]) ----
    for (int t = 0; t < Tn; ++t) {
        const int cur = t & 1, nxt = cur ^ 1;

        // prefetch x(t+1) into the next buffer while we compute on `cur`
        if (t + 1 < Tn) {
            #pragma unroll
            for (int rep = 0; rep < 2; ++rep) {
                const int idx = tid + rep * 512;
                const int r = idx >> 6, col = idx & 63;
                const float f = __builtin_nontemporal_load(
                    x + (size_t)(wg * MROW + r) * Tn * DIN
                      + (size_t)(t + 1) * DIN + col);
                hbuf[nxt][r][H + col] = (bf16_t)f;
            }
        }

        // two independent WMMA accumulation chains (shorter RAW latency chain)
        v8f acc0, acc1;
        #pragma unroll
        for (int i = 0; i < 8; ++i) { acc0[i] = bias; acc1[i] = 0.0f; }

        #pragma unroll
        for (int c = 0; c < NK; ++c) {
            V16 a;
            const bf16_t* bp_ = &hbuf[cur][nlo][c * 32 + khi];
            a.h[0] = *(const v8bf*)(bp_);        // K = 32c+khi   .. +7
            a.h[1] = *(const v8bf*)(bp_ + 16);   // K = 32c+khi+16.. +23
            if (c & 1)
                acc1 = __builtin_amdgcn_wmma_f32_16x16x32_bf16(
                           false, a.v, false, w[c].v, (short)0, acc1, false, false);
            else
                acc0 = __builtin_amdgcn_wmma_f32_16x16x32_bf16(
                           false, a.v, false, w[c].v, (short)0, acc0, false, false);
        }

        // tanh (hardware trans op, branch-free) and write h_{t+1} into the
        // next buffer (C/D layout: VGPR i, lanes 0-15 -> M=i, lanes 16-31 ->
        // M=i+8; N = lane&15).
        #pragma unroll
        for (int i = 0; i < 8; ++i) {
            const float hv = fast_tanh(acc0[i] + acc1[i]);
            const int m = i + ((lane >> 4) << 3);
            hbuf[nxt][m][wave * 16 + nlo] = (bf16_t)hv;
        }
        __syncthreads();
    }

    // ---- output projection: out = h_T @ Whp + bp  (h_T lives in hbuf[0]) ----
    const int fin = Tn & 1;                   // == 0 for T=1024
    if (wave < C / 16) {                      // 8 waves cover C=128
        const int oc = wave * 16 + nlo;
        const float bo = bp[oc];
        v8f acc;
        #pragma unroll
        for (int i = 0; i < 8; ++i) acc[i] = bo;
        #pragma unroll
        for (int c = 0; c < H / 32; ++c) {
            V16 b, a;
            #pragma unroll
            for (int e = 0; e < 16; ++e) {
                const int k = c * 32 + khi + (e < 8 ? e : (e - 8) + 16);
                b.e[e] = (bf16_t)Whp[(size_t)k * C + oc];
            }
            const bf16_t* bp_ = &hbuf[fin][nlo][c * 32 + khi];
            a.h[0] = *(const v8bf*)(bp_);
            a.h[1] = *(const v8bf*)(bp_ + 16);
            acc = __builtin_amdgcn_wmma_f32_16x16x32_bf16(
                      false, a.v, false, b.v, (short)0, acc, false, false);
        }
        #pragma unroll
        for (int i = 0; i < 8; ++i) {
            const int m = i + ((lane >> 4) << 3);
            out[(size_t)(wg * MROW + m) * C + oc] = acc[i];
        }
    }
}

extern "C" void kernel_launch(void* const* d_in, const int* in_sizes, int n_in,
                              void* d_out, int out_size, void* d_ws, size_t ws_size,
                              hipStream_t stream) {
    (void)in_sizes; (void)n_in; (void)d_ws; (void)ws_size; (void)out_size;
    const float* x   = (const float*)d_in[0];
    const float* Wxh = (const float*)d_in[1];
    const float* Whh = (const float*)d_in[2];
    const float* bh  = (const float*)d_in[3];
    const float* Whp = (const float*)d_in[4];
    const float* bp  = (const float*)d_in[5];
    float* out = (float*)d_out;

    // 32 workgroups (B/16 batch M-tiles) x 512 threads (16 waves, one N-tile each)
    VanillaRNN_fused_kernel<<<dim3(32), dim3(512), 0, stream>>>(
        x, Wxh, Whh, bh, Whp, bp, out);
}
